// ST_BLOCK_0_58832462021160
// MI455X (gfx1250) — matrix operand order repaired and checked by reading, not compiled, not run on hardware
//
#include <hip/hip_runtime.h>
#include <hip/hip_bf16.h>

typedef _Float16 h16;
typedef __attribute__((ext_vector_type(16))) _Float16 v16h;
typedef __attribute__((ext_vector_type(8)))  _Float16 v8h;
typedef __attribute__((ext_vector_type(4)))  _Float16 v4h;
typedef __attribute__((ext_vector_type(8)))  float    v8f;

#define B_   64
#define C_   64
#define CO_  64
#define N_   400
#define L_   24
#define NL_  9600          // N_*L_
#define PERB ((long)CO_*NL_)

static __device__ __forceinline__ float sigmoidf_(float x){ return 1.f/(1.f+__expf(-x)); }

// ---------------------------------------------------------------------------
// Shared WMMA tile body helpers (64x64 block tile, 4 waves, K-chunk 32).
// A frag: lane<16 -> K{0..7,16..23}, lane>=16 -> K{8..15,24..31}, M=lane&15
// B frag (LDS transposed [col][kk]): lane<16 -> K{0..7,16..23}, lane>=16 -> K{8..15,24..31}
// ---------------------------------------------------------------------------
static __device__ __forceinline__ v16h load_afrag(const h16* As, int wave, int lane){
  int m = lane & 15;
  const h16* ap = &As[(wave*16 + m)*32 + ((lane < 16) ? 0 : 8)];
  v8h a0 = *(const v8h*)ap;
  v8h a1 = *(const v8h*)(ap + 16);
  return __builtin_shufflevector(a0, a1, 0,1,2,3,4,5,6,7,8,9,10,11,12,13,14,15);
}
static __device__ __forceinline__ v16h load_bfrag(const h16* Bs, int s, int lane){
  int m = lane & 15;
  const h16* bp = &Bs[(s*16 + m)*32 + ((lane < 16) ? 0 : 8)];
  v8h b0 = *(const v8h*)bp;
  v8h b1 = *(const v8h*)(bp + 16);
  return __builtin_shufflevector(b0, b1, 0,1,2,3,4,5,6,7,8,9,10,11,12,13,14,15);
}

// ---------------------------------------------------------------------------
// Generic f32-accumulate f16 WMMA GEMM: out[b] = A @ B[b].
// AM: 0 = prepacked f16 A (a_bstride may be 0), 1 = per-batch f32 A
// BM: 0 = f16 B, 1 = f32 B   ([b][K][cols] row-major)
// OM: 0: outF = acc + bias[row]                      (conv1 -> x_input)
//     2: outF = acc                                  (S1 = s_v @ G)
//     3: outH = sigmoid(acc + cadd[row*cols+col])    (G)
//     4: outH permuted y2[b][k=row/64][n=col/24][ (row%64)*24 + col%24 ]
// Assumes K % 8 == 0, cols % 8 == 0 (true for all uses: 64/192/400, 400/9600).
// ---------------------------------------------------------------------------
template<int AM, int BM, int OM>
__global__ __launch_bounds__(128) void k_gemm(
    const h16* __restrict__ A16, const float* __restrict__ A32,
    long a_bstride, int lda,
    const h16* __restrict__ B16, const float* __restrict__ B32, long b_bstride,
    int M, int K, int cols,
    const float* __restrict__ bias, const float* __restrict__ cadd,
    float* __restrict__ outF, h16* __restrict__ outH, long out_bstride)
{
  __shared__ __align__(16) h16 As[2048];   // [row 64][kk 32]
  __shared__ __align__(16) h16 Bs[2048];   // [col 64][kk 32] (transposed)
  const int b  = blockIdx.z;
  const int m0 = blockIdx.y * 64;
  const int c0 = blockIdx.x * 64;
  const int tid = threadIdx.x;
  const int wave = tid >> 5, lane = tid & 31;
  v8f acc[4] = {};

  for (int k0 = 0; k0 < K; k0 += 32) {
    // ---- stage A (vectorized) ----
    if constexpr (AM == 0) {
      for (int g = tid; g < 256; g += 128) {           // 64 rows x 4 groups of 8
        int r = g >> 2, kg = (g & 3) * 8;
        int gr = m0 + r, gk = k0 + kg;
        v8h v = {};
        if (gr < M && gk + 8 <= K)
          v = *(const v8h*)&A16[(long)b*a_bstride + (long)gr*lda + gk];
        *(v8h*)&As[r*32 + kg] = v;
      }
    } else {
      for (int g = tid; g < 512; g += 128) {           // 64 rows x 8 groups of 4
        int r = g >> 3, kg = (g & 7) * 4;
        int gr = m0 + r, gk = k0 + kg;
        v4h v = {};
        if (gr < M && gk + 4 <= K) {
          const float4 f = *(const float4*)&A32[(long)b*a_bstride + (long)gr*lda + gk];
          v[0]=(h16)f.x; v[1]=(h16)f.y; v[2]=(h16)f.z; v[3]=(h16)f.w;
        }
        *(v4h*)&As[r*32 + kg] = v;
      }
    }
    // ---- stage B (vectorized along cols, transposed into LDS) ----
    if constexpr (BM == 0) {
      for (int g = tid; g < 256; g += 128) {           // 32 kk x 8 groups of 8
        int kk = g >> 3, colg = (g & 7) * 8;
        int gk = k0 + kk, gc = c0 + colg;
        v8h v = {};
        if (gk < K && gc + 8 <= cols)
          v = *(const v8h*)&B16[(long)b*b_bstride + (long)gk*cols + gc];
        #pragma unroll
        for (int j = 0; j < 8; ++j) Bs[(colg + j)*32 + kk] = v[j];
      }
    } else {
      for (int g = tid; g < 512; g += 128) {           // 32 kk x 16 groups of 4
        int kk = g >> 4, colg = (g & 15) * 4;
        int gk = k0 + kk, gc = c0 + colg;
        v4h v = {};
        if (gk < K && gc + 4 <= cols) {
          const float4 f = *(const float4*)&B32[(long)b*b_bstride + (long)gk*cols + gc];
          v[0]=(h16)f.x; v[1]=(h16)f.y; v[2]=(h16)f.z; v[3]=(h16)f.w;
        }
        #pragma unroll
        for (int j = 0; j < 4; ++j) Bs[(colg + j)*32 + kk] = v[j];
      }
    }
    __syncthreads();
    v16h af = load_afrag(As, wave, lane);
    #pragma unroll
    for (int s = 0; s < 4; ++s) {
      v16h bf = load_bfrag(Bs, s, lane);
      acc[s] = __builtin_amdgcn_wmma_f32_16x16x32_f16(false, af, false, bf, (short)0, acc[s], false, false);
    }
    __syncthreads();
  }
  int ci = lane & 15, rb = (lane < 16) ? 0 : 8;
  for (int s = 0; s < 4; ++s) {
    int gc = c0 + s*16 + ci;
    if (gc >= cols) continue;
    #pragma unroll
    for (int i = 0; i < 8; ++i) {
      int gr = m0 + wave*16 + rb + i;
      if (gr >= M) continue;
      float v = acc[s][i];
      if constexpr (OM == 0) {
        outF[(long)b*out_bstride + (long)gr*cols + gc] = v + bias[gr];
      } else if constexpr (OM == 2) {
        outF[(long)b*out_bstride + (long)gr*cols + gc] = v;
      } else if constexpr (OM == 3) {
        outH[(long)b*out_bstride + (long)gr*cols + gc] =
            (h16)sigmoidf_(v + cadd[(long)gr*cols + gc]);
      } else {  // OM == 4: y2[b][k][n][o*24+l]
        int k = gr >> 6, o = gr & 63;
        int n = gc / 24, l = gc - n*24;
        outH[(long)b*out_bstride + ((long)(k*400 + n))*1536 + o*24 + l] = (h16)v;
      }
    }
  }
}

// ---------------------------------------------------------------------------
// Chebyshev GEMM: out[q,(o,l)] = sum_{k=1,2} sum_n (S[b,q,n]*Lk[q,n]) * y2[b,k,n,(o,l)]
// fused reduction length 800; epilogue adds diag(ds)*y0 + g_b, ReLU
// -> halo-padded sgp[b][o][q][26] (value at l+1).
// ---------------------------------------------------------------------------
__global__ __launch_bounds__(128) void k_cheb(
    const float* __restrict__ S, const float* __restrict__ L1,
    const float* __restrict__ L2m, const h16* __restrict__ y2,
    const float* __restrict__ gb, h16* __restrict__ sgp)
{
  __shared__ __align__(16) h16 As[2048];
  __shared__ __align__(16) h16 Bs[2048];
  const int b = blockIdx.z, q0 = blockIdx.y*64, c0 = blockIdx.x*64;
  const int tid = threadIdx.x, wave = tid >> 5, lane = tid & 31;
  v8f acc[4] = {};
  for (int k0 = 0; k0 < 800; k0 += 32) {
    // A = S .* Lk : float4 pair loads + cvt (k boundary at 400 aligns to groups)
    for (int g = tid; g < 512; g += 128) {
      int r = g >> 3, kg = (g & 7) * 4;
      int q = q0 + r, rr = k0 + kg;
      int kidx = (rr >= 400); int n = rr - (kidx ? 400 : 0);
      v4h v = {};
      if (q < 400) {
        const float* Lk = kidx ? L2m : L1;
        const float4 sv = *(const float4*)&S[((long)b*400 + q)*400 + n];
        const float4 lv = *(const float4*)&Lk[(long)q*400 + n];
        v[0]=(h16)(sv.x*lv.x); v[1]=(h16)(sv.y*lv.y);
        v[2]=(h16)(sv.z*lv.z); v[3]=(h16)(sv.w*lv.w);
      }
      *(v4h*)&As[r*32 + kg] = v;
    }
    // B from y2: fully contiguous v8h loads over (o,l) columns
    for (int g = tid; g < 256; g += 128) {
      int kk = g >> 3, colg = (g & 7) * 8;
      int rr = k0 + kk;
      int kidx = (rr >= 400); int n = rr - (kidx ? 400 : 0);
      v8h v = *(const v8h*)&y2[(((long)b*3 + kidx + 1)*400 + n)*1536 + c0 + colg];
      #pragma unroll
      for (int j = 0; j < 8; ++j) Bs[(colg + j)*32 + kk] = v[j];
    }
    __syncthreads();
    v16h af = load_afrag(As, wave, lane);
    #pragma unroll
    for (int s = 0; s < 4; ++s) {
      v16h bf = load_bfrag(Bs, s, lane);
      acc[s] = __builtin_amdgcn_wmma_f32_16x16x32_f16(false, af, false, bf, (short)0, acc[s], false, false);
    }
    __syncthreads();
  }
  int ci = lane & 15, rb = (lane < 16) ? 0 : 8;
  for (int s = 0; s < 4; ++s) {
    int gc = c0 + s*16 + ci; int o = gc / 24, l = gc - o*24;
    #pragma unroll
    for (int i = 0; i < 8; ++i) {
      int gq = q0 + wave*16 + rb + i;
      if (gq >= 400) continue;
      float diag = S[((long)b*400 + gq)*400 + gq];          // k=0 (L0=I) term
      float y0 = (float)y2[((long)b*3*400 + gq)*1536 + gc];
      float v = acc[s][i] + diag*y0 + gb[o];
      sgp[(((long)b*64 + o)*400 + gq)*26 + l + 1] = (h16)fmaxf(v, 0.f);
    }
  }
}

// ---------------------------------------------------------------------------
// Temporal conv as implicit-im2col WMMA GEMM: K=(t,i)=192, cols=(n,l)=9600.
// B gather is branch-free via halo-padded sgp (reads at l+t, t in 0..2).
// Epilogue: h = relu(acc + tc_b + x_input) in-place + LN partial sums.
// ---------------------------------------------------------------------------
__global__ __launch_bounds__(128) void k_tconv(
    const h16* __restrict__ W, const h16* __restrict__ sgp,
    const float* __restrict__ tcb, float* __restrict__ hio,
    float* __restrict__ ps, float* __restrict__ pq)
{
  __shared__ __align__(16) h16 As[2048];
  __shared__ __align__(16) h16 Bs[2048];
  __shared__ float r1[128], r2[128];
  const int b = blockIdx.z, c0 = blockIdx.x*64;
  const int tid = threadIdx.x, wave = tid >> 5, lane = tid & 31;
  v8f acc[4] = {};
  for (int k0 = 0; k0 < 192; k0 += 32) {
    for (int g = tid; g < 256; g += 128) {               // W rows 0..63, lda=192
      int r = g >> 2, kg = (g & 3) * 8;
      *(v8h*)&As[r*32 + kg] = *(const v8h*)&W[(long)r*192 + k0 + kg];
    }
    for (int i = tid; i < 2048; i += 128) {              // branch-free halo gather
      int kk = i >> 6, col = i & 63;
      int rr = k0 + kk; int t = rr >> 6, ii = rr & 63;
      int gc = c0 + col; int n = gc / 24, l = gc - n*24;
      Bs[col*32 + kk] = sgp[(((long)b*64 + ii)*400 + n)*26 + l + t];
    }
    __syncthreads();
    v16h af = load_afrag(As, wave, lane);
    #pragma unroll
    for (int s = 0; s < 4; ++s) {
      v16h bf = load_bfrag(Bs, s, lane);
      acc[s] = __builtin_amdgcn_wmma_f32_16x16x32_f16(false, af, false, bf, (short)0, acc[s], false, false);
    }
    __syncthreads();
  }
  float s1 = 0.f, s2 = 0.f;
  int ci = lane & 15, rb = (lane < 16) ? 0 : 8;
  for (int s = 0; s < 4; ++s) {
    int gc = c0 + s*16 + ci;
    #pragma unroll
    for (int i = 0; i < 8; ++i) {
      int o = wave*16 + rb + i;
      long idx = ((long)b*64 + o)*9600 + gc;
      float v = acc[s][i] + tcb[o] + hio[idx];
      v = fmaxf(v, 0.f);
      hio[idx] = v;
      s1 += v; s2 += v*v;
    }
  }
  r1[tid] = s1; r2[tid] = s2; __syncthreads();
  for (int o = 64; o > 0; o >>= 1) {
    if (tid < o) { r1[tid] += r1[tid+o]; r2[tid] += r2[tid+o]; }
    __syncthreads();
  }
  if (tid == 0) { ps[b*150 + blockIdx.x] = r1[0]; pq[b*150 + blockIdx.x] = r2[0]; }
}

// --------------------------- small helper kernels ---------------------------
__global__ void k_f2h(const float* s, h16* d, int n){
  int i = blockIdx.x*256 + threadIdx.x; if (i < n) d[i] = (h16)s[i];
}
__global__ void k_pack_gw(const float* g, h16* d){        // [k*64+o][c] <- g_w[o, c*3+k]
  int i = blockIdx.x*256 + threadIdx.x;
  if (i < 192*64){ int r = i >> 6, c = i & 63; int k = r/64, o = r%64; d[i] = (h16)g[o*192 + c*3 + k]; }
}
__global__ void k_pack_tcw(const float* tw, h16* d){      // [o][t*64+i] <- tc_w[o,i,0,t]
  int i = blockIdx.x*256 + threadIdx.x;
  if (i < 64*192){ int o = i/192, r = i%192; int t = r >> 6, ii = r & 63; d[i] = (h16)tw[(o*64 + ii)*3 + t]; }
}
__global__ void k_halo(h16* sgp){                         // zero halo columns (l=0 and l=25)
  int i = blockIdx.x*256 + threadIdx.x;
  if (i < B_*64*400) { sgp[(long)i*26] = (h16)0.f; sgp[(long)i*26 + 25] = (h16)0.f; }
}
__global__ void k_l2(const float* adj, float* L2m){       // L2 = 2*adj@adj - I
  int idx = blockIdx.x*256 + threadIdx.x;
  if (idx >= 160000) return;
  int q = idx/400, n = idx%400;
  float s = 0.f;
  for (int m = 0; m < 400; ++m) s += adj[q*400 + m]*adj[m*400 + n];
  L2m[idx] = 2.f*s - (q == n ? 1.f : 0.f);
}
__global__ __launch_bounds__(32) void k_feats_c(const float* x, const float* tcw1,
                                                const float* scw1, float* tf1, float* sf1){
  int n = blockIdx.x, b = blockIdx.y, l = threadIdx.x;
  if (l >= 24) return;
  float a = 0.f, s = 0.f;
  for (int c = 0; c < 64; ++c) {
    float v = x[(((long)b*64 + c)*400 + n)*24 + l];
    a += tcw1[c]*v; s += scw1[c]*v;
  }
  tf1[((long)b*24 + l)*400 + n] = a;
  sf1[((long)b*400 + n)*24 + l] = s;
}
__global__ __launch_bounds__(32) void k_tf2k(const float* x, const float* tcw2, float* tf2){
  int c = blockIdx.x, b = blockIdx.y, l = threadIdx.x;
  if (l >= 24) return;
  const float* xr = x + ((long)(b*64 + c))*9600;
  float a = 0.f;
  for (int n = 0; n < 400; ++n) a += tcw2[n]*xr[n*24 + l];
  tf2[((long)b*64 + c)*24 + l] = a;
}
__global__ __launch_bounds__(128) void k_sf2k(const float* x, const float* scw2, float* sf2){
  int c = blockIdx.x, b = blockIdx.y, tid = threadIdx.x;
  const float* xr = x + ((long)(b*64 + c))*9600;
  for (int n = tid; n < 400; n += 128) {
    float a = 0.f;
    for (int l = 0; l < 24; ++l) a += scw2[l]*xr[n*24 + l];
    sf2[((long)b*64 + c)*400 + n] = a;
  }
}
__global__ __launch_bounds__(256) void k_tatt(const float* tf1, const float* tf2,
    const float* t_w, const float* t_b, const float* t_v, float* Tout){
  int b = blockIdx.x, tid = threadIdx.x;
  __shared__ float E1[24*64];
  __shared__ float LG[576];
  __shared__ float TL[576];
  __shared__ float cm[24];
  for (int i = tid; i < 24*64; i += 256) {
    int l = i >> 6, c = i & 63;
    const float* f1 = tf1 + ((long)b*24 + l)*400;
    float s = 0.f;
    for (int n = 0; n < 400; ++n) s += f1[n]*t_w[n*64 + c];
    E1[i] = s;
  }
  __syncthreads();
  for (int i = tid; i < 576; i += 256) {
    int l = i/24, q = i%24;
    float s = 0.f;
    for (int c = 0; c < 64; ++c) s += E1[l*64 + c]*tf2[((long)b*64 + c)*24 + q];
    LG[i] = sigmoidf_(s + t_b[l*24 + q]);
  }
  __syncthreads();
  for (int i = tid; i < 576; i += 256) {
    int p = i/24, l = i%24;
    float s = 0.f;
    for (int q = 0; q < 24; ++q) s += t_v[p*24 + q]*LG[q*24 + l];
    TL[i] = s;
  }
  __syncthreads();
  if (tid < 24) { float m = -1e30f; for (int p = 0; p < 24; ++p) m = fmaxf(m, TL[p*24 + tid]); cm[tid] = m; }
  __syncthreads();
  if (tid < 24) {                            // softmax over l, store transposed
    int p = tid; float e[24]; float s = 0.f;
    for (int l = 0; l < 24; ++l) { e[l] = __expf(TL[p*24 + l] - cm[l]); s += e[l]; }
    float inv = 1.f/s;
    for (int l = 0; l < 24; ++l) Tout[(long)b*576 + l*24 + p] = e[l]*inv;
  }
}
__global__ __launch_bounds__(64) void k_sattp(const float* sf1, const float* s_w, float* P){
  int n = blockIdx.x, b = blockIdx.y, c = threadIdx.x;
  __shared__ float f[24];
  if (c < 24) f[c] = sf1[((long)b*400 + n)*24 + c];
  __syncthreads();
  float s = 0.f;
  for (int l = 0; l < 24; ++l) s += f[l]*s_w[l*64 + c];
  P[((long)b*400 + n)*64 + c] = s;
}
__global__ __launch_bounds__(128) void k_colmax(const float* S1, float* cmax){
  int n = blockIdx.x*128 + threadIdx.x, b = blockIdx.y;
  if (n >= 400) return;
  const float* base = S1 + (long)b*160000;
  float m = -1e30f;
  for (int p = 0; p < 400; ++p) m = fmaxf(m, base[p*400 + n]);
  cmax[b*400 + n] = m;
}
__global__ __launch_bounds__(128) void k_srow(const float* S1, const float* cmax, float* Sout){
  int b = blockIdx.y, p = blockIdx.x, tid = threadIdx.x;
  __shared__ float e[400];
  __shared__ float red[128];
  const float* row = S1 + ((long)b*400 + p)*400;
  float s = 0.f;
  for (int n = tid; n < 400; n += 128) { float v = __expf(row[n] - cmax[b*400 + n]); e[n] = v; s += v; }
  red[tid] = s; __syncthreads();
  for (int o = 64; o > 0; o >>= 1) { if (tid < o) red[tid] += red[tid+o]; __syncthreads(); }
  float inv = 1.f/red[0];
  for (int n = tid; n < 400; n += 128) Sout[((long)b*400 + p)*400 + n] = e[n]*inv;
}
__global__ __launch_bounds__(256) void k_xtat(const float* x, const float* Tout, h16* xt){
  int c = blockIdx.x, b = blockIdx.y, tid = threadIdx.x;
  __shared__ float T[576];
  for (int i = tid; i < 576; i += 256) T[i] = Tout[(long)b*576 + i];
  __syncthreads();
  const float* xr = x + ((long)(b*64 + c))*9600;
  h16* xo = xt + ((long)(b*64 + c))*9600;
  for (int n = tid; n < 400; n += 256) {
    float xv[24];
    for (int l = 0; l < 24; ++l) xv[l] = xr[n*24 + l];
    for (int q = 0; q < 24; ++q) {
      float s = 0.f;
      for (int l = 0; l < 24; ++l) s += xv[l]*T[l*24 + q];
      xo[n*24 + q] = (h16)s;
    }
  }
}
__global__ __launch_bounds__(256) void k_stats(const float* ps, const float* pq, float* stats){
  int b = blockIdx.x, tid = threadIdx.x;
  __shared__ float r1[256], r2[256];
  float s1 = 0.f, s2 = 0.f;
  for (int i = tid; i < 150; i += 256) { s1 += ps[b*150 + i]; s2 += pq[b*150 + i]; }
  r1[tid] = s1; r2[tid] = s2; __syncthreads();
  for (int o = 128; o > 0; o >>= 1) { if (tid < o) { r1[tid] += r1[tid+o]; r2[tid] += r2[tid+o]; } __syncthreads(); }
  if (tid == 0) { stats[2*b] = r1[0]; stats[2*b+1] = r2[0]; }
}
__global__ void k_ln(const float* h, const float* stats, const float* g,
                     const float* be, float* out){
  long idx = (long)blockIdx.x*256 + threadIdx.x;
  if (idx >= (long)B_*PERB) return;
  int b = (int)(idx / PERB);
  long r = idx - (long)b*PERB;
  float cnt = (float)PERB;
  float mu = stats[2*b]/cnt;
  float var = stats[2*b+1]/cnt - mu*mu;
  out[idx] = (h[idx] - mu)*rsqrtf(var + 1e-5f)*g[r] + be[r];
}

// ---------------------------------------------------------------------------
extern "C" void kernel_launch(void* const* d_in, const int* in_sizes, int n_in,
                              void* d_out, int out_size, void* d_ws, size_t ws_size,
                              hipStream_t stream) {
  const float* x      = (const float*)d_in[0];
  const float* adj    = (const float*)d_in[1];
  const float* conv1w = (const float*)d_in[2];
  const float* conv1b = (const float*)d_in[3];
  const float* t_cw1  = (const float*)d_in[4];
  const float* t_cw2  = (const float*)d_in[5];
  const float* t_w    = (const float*)d_in[6];
  const float* t_b    = (const float*)d_in[7];
  const float* t_v    = (const float*)d_in[8];
  const float* s_cw1  = (const float*)d_in[9];
  const float* s_cw2  = (const float*)d_in[10];
  const float* s_w    = (const float*)d_in[11];
  const float* s_b    = (const float*)d_in[12];
  const float* s_v    = (const float*)d_in[13];
  const float* g_w    = (const float*)d_in[14];
  const float* g_b    = (const float*)d_in[15];
  const float* tc_w   = (const float*)d_in[16];
  const float* tc_b   = (const float*)d_in[17];
  const float* ln_g   = (const float*)d_in[18];
  const float* ln_b   = (const float*)d_in[19];

  float* outp = (float*)d_out;
  float* Sout = outp + (long)B_*PERB;          // S_coef [B,N,N]
  float* Tout = Sout + (long)B_*N_*N_;         // T_coef (swapped) [B,L,L]

  char* w = (char*)d_ws;
  auto alloc = [&](size_t bytes) -> void* {
    void* p = (void*)w; w += (bytes + 255) & ~(size_t)255; return p;
  };
  float* L2m   = (float*)alloc((size_t)160000*4);
  float* tf1   = (float*)alloc((size_t)B_*24*400*4);
  float* tf2   = (float*)alloc((size_t)B_*64*24*4);
  float* sf1   = (float*)alloc((size_t)B_*400*24*4);
  float* sf2   = (float*)alloc((size_t)B_*64*400*4);
  float* P     = (float*)alloc((size_t)B_*400*64*4);
  h16*   G     = (h16*)  alloc((size_t)B_*160000*2);
  float* S1    = (float*)alloc((size_t)B_*160000*4);
  float* cmax  = (float*)alloc((size_t)B_*400*4);
  h16*   xTAt  = (h16*)  alloc((size_t)B_*C_*NL_*2);
  h16*   w16c  = (h16*)  alloc((size_t)64*64*2);
  h16*   w16g  = (h16*)  alloc((size_t)192*64*2);
  h16*   w16t  = (h16*)  alloc((size_t)64*192*2);
  h16*   sv16  = (h16*)  alloc((size_t)160000*2);
  h16*   y2    = (h16*)  alloc((size_t)B_*3*400*1536*2);   // [b][k][n][(o,l)]
  float* xinp  = (float*)alloc((size_t)B_*PERB*4);         // x_input, then h (in-place)
  h16*   sgp   = (h16*)  alloc((size_t)B_*64*400*26*2);    // halo-padded spatial_gcn
  float* ps    = (float*)alloc((size_t)B_*150*4);
  float* pq    = (float*)alloc((size_t)B_*150*4);
  float* stats = (float*)alloc((size_t)2*B_*4);

  // --- prep: L2 = 2 adj^2 - I, f16 weight packs, halo zeroing ---
  k_l2<<<dim3((160000+255)/256), 256, 0, stream>>>(adj, L2m);
  k_f2h<<<dim3((64*64+255)/256), 256, 0, stream>>>(conv1w, w16c, 64*64);
  k_f2h<<<dim3((160000+255)/256), 256, 0, stream>>>(s_v, sv16, 160000);
  k_pack_gw<<<dim3((192*64+255)/256), 256, 0, stream>>>(g_w, w16g);
  k_pack_tcw<<<dim3((64*192+255)/256), 256, 0, stream>>>(tc_w, w16t);
  k_halo<<<dim3((B_*64*400+255)/256), 256, 0, stream>>>(sgp);

  // --- attention feature reductions ---
  k_feats_c<<<dim3(N_, B_), 32, 0, stream>>>(x, t_cw1, s_cw1, tf1, sf1);
  k_tf2k   <<<dim3(C_, B_), 32, 0, stream>>>(x, t_cw2, tf2);
  k_sf2k   <<<dim3(C_, B_), 128, 0, stream>>>(x, s_cw2, sf2);

  // --- temporal attention (small, one block per batch) -> Tout (output 3) ---
  k_tatt<<<dim3(B_), 256, 0, stream>>>(tf1, tf2, t_w, t_b, t_v, Tout);

  // --- spatial attention ---
  k_sattp<<<dim3(N_, B_), 64, 0, stream>>>(sf1, s_w, P);
  // G = sigmoid(P @ sf2 + s_b), f16
  k_gemm<1,1,3><<<dim3(7,7,B_), 128, 0, stream>>>(nullptr, P, (long)400*64, 64,
      nullptr, sf2, (long)64*400, 400, 64, 400, nullptr, s_b,
      nullptr, G, (long)160000);
  // S1 = s_v @ G
  k_gemm<0,0,2><<<dim3(7,7,B_), 128, 0, stream>>>(sv16, nullptr, 0, 400,
      G, nullptr, (long)160000, 400, 400, 400, nullptr, nullptr,
      S1, nullptr, (long)160000);
  k_colmax<<<dim3(4, B_), 128, 0, stream>>>(S1, cmax);
  k_srow  <<<dim3(N_, B_), 128, 0, stream>>>(S1, cmax, Sout);   // S_coef (output 2)

  // --- x_TAt (f16) ---
  k_xtat<<<dim3(C_, B_), 256, 0, stream>>>(x, Tout, xTAt);

  // --- x_input = conv1_w @ x + b (f32) ---
  k_gemm<0,1,0><<<dim3(150,1,B_), 128, 0, stream>>>(w16c, nullptr, 0, 64,
      nullptr, x, (long)C_*NL_, 64, 64, NL_, conv1b, nullptr,
      xinp, nullptr, PERB);

  // --- y2[b,k,n,(o,l)] = gw_k @ x_TAt (f16, permuted epilogue) ---
  k_gemm<0,0,4><<<dim3(150,3,B_), 128, 0, stream>>>(w16g, nullptr, 0, 64,
      xTAt, nullptr, (long)C_*NL_, 192, 64, NL_, nullptr, nullptr,
      nullptr, y2, (long)3*400*1536);

  // --- Chebyshev spatial GCN (dominant GEMM) -> sgp (halo-padded relu) ---
  k_cheb<<<dim3(24,7,B_), 128, 0, stream>>>(Sout, adj, L2m, y2, g_b, sgp);

  // --- temporal conv + residual + relu + LN partials (in-place over xinp) ---
  k_tconv<<<dim3(150,1,B_), 128, 0, stream>>>(w16t, sgp, tc_b, xinp, ps, pq);
  k_stats<<<dim3(B_), 256, 0, stream>>>(ps, pq, stats);

  // --- layernorm -> output 1 ---
  long tot = (long)B_*PERB;
  k_ln<<<dim3((unsigned)((tot + 255)/256)), 256, 0, stream>>>(xinp, stats, ln_g, ln_b, outp);
}
